// AggregationMPNN_76768245449269
// MI455X (gfx1250) — matrix operand order repaired and checked by reading, not compiled, not run on hardware
//
#include <hip/hip_runtime.h>

typedef __attribute__((ext_vector_type(16))) _Float16 v16h;
typedef __attribute__((ext_vector_type(8)))  float    v8f;
typedef __attribute__((ext_vector_type(4)))  unsigned int u32x4;

#define BIG_NEG (-1.0e6f)

constexpr int GN  = 64;    // graphs
constexpr int GV  = 64;    // nodes per graph
constexpr int GE  = 8;     // edge types
constexpr int NF  = 64;    // node features
constexpr int H   = 256;   // hidden
constexpr int M   = 128;   // message dim
constexpr int OUTD = 256;  // output dim
constexpr int NV  = GN * GV;   // 4096 rows
constexpr int CAT = H + NF;    // 320
constexpr int H3  = 3 * H;     // 768
constexpr int EM  = GE * M;    // 1024

__device__ __forceinline__ float sigmoidf_(float x) { return 1.0f / (1.0f + expf(-x)); }

// ---------------------------------------------------------------------------
// Swizzle an f32 weight matrix B[K x Cols] into WMMA B-fragment tile order,
// converted to f16.  Tile (kb, cb) lives at halfword offset (cb*Kt + kb)*512;
// inside a tile, lane l's 16 halves are contiguous at l*16.
// Source indexing: B[k, col] = src[k*strideK + (col/Minner)*strideCouter
//                                  + (col%Minner)*strideCinner]
// ---------------------------------------------------------------------------
__global__ void swizzle_b_kernel(const float* __restrict__ src, _Float16* __restrict__ dst,
                                 int K, int Cols, int strideK, int strideCouter,
                                 int Minner, int strideCinner) {
  int idx = blockIdx.x * blockDim.x + threadIdx.x;
  int total = K * Cols;
  if (idx >= total) return;
  int Kt   = K >> 5;
  int t    = idx >> 9;
  int lane = (idx >> 4) & 31;
  int j    = idx & 15;
  int kb   = t % Kt;
  int cb   = t / Kt;
  int col  = cb * 16 + (lane & 15);
  int k    = kb * 32 + ((j < 8) ? 0 : 16) + ((lane < 16) ? 0 : 8) + (j & 7);
  int co   = col / Minner;
  int ci   = col % Minner;
  size_t s = (size_t)k * strideK + (size_t)co * strideCouter + (size_t)ci * strideCinner;
  dst[idx] = (_Float16)src[s];
}

// ---------------------------------------------------------------------------
// A-fragment load (16x32 f16, row-major source), per ISA 7.12.2:
//   lane 0-15 : row = lane,    K = {0..7} in h[0..7],  {16..23} in h[8..15]
//   lane16-31 : row = lane-16, K = {8..15} in h[0..7], {24..31} in h[8..15]
// ---------------------------------------------------------------------------
__device__ __forceinline__ v16h load_a_frag(const _Float16* __restrict__ A, int lda,
                                            int rowBase, int kblock, int lane) {
  int row   = rowBase + (lane & 15);
  int kbase = (lane < 16) ? 0 : 8;
  const _Float16* p = A + (size_t)row * lda + kblock + kbase;
  union { v16h h; u32x4 u[2]; } f;
  f.u[0] = *(const u32x4*)(p);
  f.u[1] = *(const u32x4*)(p + 16);
  return f.h;
}

// ---------------------------------------------------------------------------
// Each wave computes a 16 x (16*CT) output strip: one A-fragment feeds CT
// WMMAs per K-step (A-load amortization; 4 accumulators live in VGPRs).
// C = A(f16) x B(f16) + bias[col], f32 accumulate, store f32 or f16.
// ---------------------------------------------------------------------------
template <typename OutT, int CT>
__global__ __launch_bounds__(256)
void gemm_wmma_kernel(const _Float16* __restrict__ A, int lda,
                      const _Float16* __restrict__ Bsw,
                      const float* __restrict__ bias,
                      OutT* __restrict__ C, int ldc,
                      int Kt, int rowTiles) {
  int wave = blockIdx.x * (blockDim.x >> 5) + (threadIdx.x >> 5);
  int lane = threadIdx.x & 31;
  int rt = wave % rowTiles;          // row tile (fastest: waves in a block share B)
  int cg = wave / rowTiles;          // group of CT column tiles
  int rowBase = rt * 16;
  const _Float16* bp = Bsw + (size_t)(cg * CT) * Kt * 512 + lane * 16;
  v8f acc[CT] = {};
  for (int kb = 0; kb < Kt; ++kb) {
    v16h a = load_a_frag(A, lda, rowBase, kb * 32, lane);
#pragma unroll
    for (int i = 0; i < CT; ++i) {
      union { v16h h; u32x4 u[2]; } bf;
      const _Float16* bpi = bp + (size_t)i * Kt * 512;
      bf.u[0] = *(const u32x4*)(bpi);
      bf.u[1] = *(const u32x4*)(bpi + 8);
      acc[i] = __builtin_amdgcn_wmma_f32_16x16x32_f16(false, a, false, bf.h,
                                                      (short)0, acc[i], false, false);
    }
    bp += 512;
  }
  int rowOff = rowBase + ((lane < 16) ? 0 : 8);
#pragma unroll
  for (int i = 0; i < CT; ++i) {
    int col = (cg * CT + i) * 16 + (lane & 15);
    float b = bias[col];
#pragma unroll
    for (int r = 0; r < 8; ++r) {
      C[(size_t)(rowOff + r) * ldc + col] = (OutT)(acc[i][r] + b);
    }
  }
}

// ---------------------------------------------------------------------------
// hidden f32 = zero-pad(nodes); catA f16 = [hidden(256) | nodes(64)]
// ---------------------------------------------------------------------------
__global__ void init_state_kernel(const float* __restrict__ nodes,
                                  float* __restrict__ hidden,
                                  _Float16* __restrict__ catA) {
  int idx = blockIdx.x * blockDim.x + threadIdx.x;
  if (idx >= NV * CAT) return;
  int row = idx / CAT;
  int c   = idx - row * CAT;
  float v;
  if (c < H) {
    v = (c < NF) ? nodes[(size_t)row * NF + c] : 0.0f;
    hidden[(size_t)row * H + c] = v;
  } else {
    v = nodes[(size_t)row * NF + (c - H)];
  }
  catA[idx] = (_Float16)v;
}

// One-hot [N,V,V,E] -> edge-type byte map (0xFF = no edge)
__global__ void edge_init_kernel(const float* __restrict__ edges,
                                 unsigned char* __restrict__ et) {
  int idx = blockIdx.x * blockDim.x + threadIdx.x;
  if (idx >= GN * GV * GV) return;
  const float* e = edges + (size_t)idx * GE;
  unsigned char t = 0xFF;
#pragma unroll
  for (int k = 0; k < GE; ++k)
    if (e[k] > 0.0f) t = (unsigned char)k;
  et[idx] = t;
}

__global__ void node_mask_kernel(const unsigned char* __restrict__ et,
                                 unsigned char* __restrict__ nmask) {
  int nv = blockIdx.x * blockDim.x + threadIdx.x;
  if (nv >= NV) return;
  const unsigned char* r = et + (size_t)nv * GV;
  unsigned char m = 0;
  for (int w = 0; w < GV; ++w) m |= (unsigned char)(r[w] != 0xFF);
  nmask[nv] = m;
}

// ---------------------------------------------------------------------------
// Per (n,v): gather energies/messages by edge type, softmax over neighbours,
// weighted sum -> message[nv, m] (f16).  Edge-type row staged in LDS once;
// gathers are coalesced across m (contiguous 256B rows of att_p/msg_p).
// ---------------------------------------------------------------------------
__global__ __launch_bounds__(128)
void attn_message_kernel(const _Float16* __restrict__ msg_p,
                         const _Float16* __restrict__ att_p,
                         const unsigned char* __restrict__ et,
                         _Float16* __restrict__ message) {
  __shared__ unsigned char sE[GV];
  int nv = blockIdx.x;           // n*64 + v
  int m  = threadIdx.x;          // 0..127
  int n  = nv >> 6;
  if (threadIdx.x < GV) sE[threadIdx.x] = et[(size_t)nv * GV + threadIdx.x];
  __syncthreads();

  float mx = BIG_NEG;
  for (int w = 0; w < GV; ++w) {
    unsigned char e = sE[w];
    float en = BIG_NEG;
    if (e != 0xFF)
      en = (float)att_p[(((size_t)(n * GV + w)) * GE + e) * M + m];
    mx = fmaxf(mx, en);
  }
  float s = 0.0f, acc = 0.0f;
  for (int w = 0; w < GV; ++w) {
    unsigned char e = sE[w];
    float en = BIG_NEG, mg = 0.0f;
    if (e != 0xFF) {
      size_t base = (((size_t)(n * GV + w)) * GE + e) * M + m;
      en = (float)att_p[base];
      mg = (float)msg_p[base];
    }
    float ex = expf(en - mx);
    s += ex;
    acc += ex * mg;
  }
  message[(size_t)nv * M + m] = (_Float16)(acc / s);
}

// ---------------------------------------------------------------------------
// GRUCell elementwise update; masked by node degree; refreshes f16 catA.
// ---------------------------------------------------------------------------
__global__ void gru_update_kernel(const float* __restrict__ gi,
                                  const float* __restrict__ gh,
                                  float* __restrict__ hidden,
                                  _Float16* __restrict__ catA,
                                  const unsigned char* __restrict__ nmask) {
  int idx = blockIdx.x * blockDim.x + threadIdx.x;
  if (idx >= NV * H) return;
  int row = idx >> 8;
  int h   = idx & (H - 1);
  if (!nmask[row]) return;
  const float* a = gi + (size_t)row * H3;
  const float* b = gh + (size_t)row * H3;
  float r  = sigmoidf_(a[h] + b[h]);
  float z  = sigmoidf_(a[H + h] + b[H + h]);
  float nn = tanhf(a[2 * H + h] + r * b[2 * H + h]);
  float hv = hidden[idx];
  float hn = (1.0f - z) * nn + z * hv;
  hidden[idx] = hn;
  catA[(size_t)row * CAT + h] = (_Float16)hn;
}

// Gated masked sum over nodes -> [N, OUT]
__global__ __launch_bounds__(256)
void readout_kernel(const float* __restrict__ gl,
                    const float* __restrict__ embp,
                    const unsigned char* __restrict__ nmask,
                    float* __restrict__ out) {
  int n = blockIdx.x;
  int c = threadIdx.x;
  float s = 0.0f;
  for (int v = 0; v < GV; ++v) {
    int row = n * GV + v;
    if (nmask[row]) {
      float g = sigmoidf_(gl[(size_t)row * OUTD + c]);
      s += g * embp[(size_t)row * OUTD + c];
    }
  }
  out[(size_t)n * OUTD + c] = s;
}

// ---------------------------------------------------------------------------
extern "C" void kernel_launch(void* const* d_in, const int* in_sizes, int n_in,
                              void* d_out, int out_size, void* d_ws, size_t ws_size,
                              hipStream_t stream) {
  (void)in_sizes; (void)n_in; (void)out_size; (void)ws_size;
  const float* nodes = (const float*)d_in[0];
  const float* edges = (const float*)d_in[1];
  const float* msg_W = (const float*)d_in[2];
  const float* msg_b = (const float*)d_in[3];
  const float* att_W = (const float*)d_in[4];
  const float* att_b = (const float*)d_in[5];
  const float* W_ih  = (const float*)d_in[6];
  const float* W_hh  = (const float*)d_in[7];
  const float* b_ih  = (const float*)d_in[8];
  const float* b_hh  = (const float*)d_in[9];
  const float* gaW   = (const float*)d_in[10];
  const float* gab   = (const float*)d_in[11];
  const float* geW   = (const float*)d_in[12];
  const float* geb   = (const float*)d_in[13];
  float* out = (float*)d_out;

  char* p = (char*)d_ws;
  auto carve = [&](size_t bytes) -> char* {
    char* r = p;
    p += (bytes + 255) & ~(size_t)255;
    return r;
  };
  _Float16* Bmsg  = (_Float16*)carve((size_t)H * EM * 2);     // 512 KB
  _Float16* Batt  = (_Float16*)carve((size_t)H * EM * 2);     // 512 KB
  _Float16* Bih   = (_Float16*)carve((size_t)M * H3 * 2);     // 192 KB
  _Float16* Bhh   = (_Float16*)carve((size_t)H * H3 * 2);     // 384 KB
  _Float16* Bgatt = (_Float16*)carve((size_t)CAT * OUTD * 2); // 160 KB
  _Float16* Bgemb = (_Float16*)carve((size_t)H * OUTD * 2);   // 128 KB
  float*    hidden = (float*)carve((size_t)NV * H * 4);       // 4 MB
  _Float16* catA   = (_Float16*)carve((size_t)NV * CAT * 2);  // 2.5 MB
  unsigned char* et    = (unsigned char*)carve((size_t)GN * GV * GV); // 256 KB
  unsigned char* nmask = (unsigned char*)carve((size_t)NV);
  _Float16* msgp  = (_Float16*)carve((size_t)NV * EM * 2);    // 8 MB
  _Float16* attp  = (_Float16*)carve((size_t)NV * EM * 2);    // 8 MB
  _Float16* msgf  = (_Float16*)carve((size_t)NV * M * 2);     // 1 MB
  float*    gi    = (float*)carve((size_t)NV * H3 * 4);       // 12 MB
  float*    gh    = (float*)carve((size_t)NV * H3 * 4);       // 12 MB
  float*    gl    = (float*)carve((size_t)NV * OUTD * 4);     // 4 MB
  float*    embp  = (float*)carve((size_t)NV * OUTD * 4);     // 4 MB

  auto blocks = [](int total, int tpb) { return (total + tpb - 1) / tpb; };

  // --- weight swizzles into WMMA B-fragment tile order ---
  swizzle_b_kernel<<<blocks(H * EM, 256), 256, 0, stream>>>(
      msg_W, Bmsg, H, EM, M, H * M, M, 1);                 // [E,H,M] -> [H, E*M]
  swizzle_b_kernel<<<blocks(H * EM, 256), 256, 0, stream>>>(
      att_W, Batt, H, EM, M, H * M, M, 1);
  swizzle_b_kernel<<<blocks(M * H3, 256), 256, 0, stream>>>(
      W_ih, Bih, M, H3, 1, 0, H3, M);                      // [3H,M]^T
  swizzle_b_kernel<<<blocks(H * H3, 256), 256, 0, stream>>>(
      W_hh, Bhh, H, H3, 1, 0, H3, H);                      // [3H,H]^T
  swizzle_b_kernel<<<blocks(CAT * OUTD, 256), 256, 0, stream>>>(
      gaW, Bgatt, CAT, OUTD, OUTD, 0, OUTD, 1);            // [CAT,OUT]
  swizzle_b_kernel<<<blocks(H * OUTD, 256), 256, 0, stream>>>(
      geW, Bgemb, H, OUTD, OUTD, 0, OUTD, 1);              // [H,OUT]

  // --- state / graph-structure init ---
  init_state_kernel<<<blocks(NV * CAT, 256), 256, 0, stream>>>(nodes, hidden, catA);
  edge_init_kernel<<<blocks(GN * GV * GV, 256), 256, 0, stream>>>(edges, et);
  node_mask_kernel<<<blocks(NV, 256), 256, 0, stream>>>(et, nmask);

  constexpr int ROWT = NV / 16;  // 256 row tiles
  constexpr int CT   = 4;        // 16x64 strip per wave
  // --- 3 message passes ---
  for (int pass = 0; pass < 3; ++pass) {
    // msg_p / att_p : [NV,256] x [256,1024] -> f16
    gemm_wmma_kernel<_Float16, CT><<<ROWT * (EM / 16 / CT) / 8, 256, 0, stream>>>(
        catA, CAT, Bmsg, msg_b, msgp, EM, H / 32, ROWT);
    gemm_wmma_kernel<_Float16, CT><<<ROWT * (EM / 16 / CT) / 8, 256, 0, stream>>>(
        catA, CAT, Batt, att_b, attp, EM, H / 32, ROWT);
    // neighbourhood attention + aggregation
    attn_message_kernel<<<NV, 128, 0, stream>>>(msgp, attp, et, msgf);
    // gi = message @ W_ih^T + b_ih : [NV,128] x [128,768]
    gemm_wmma_kernel<float, CT><<<ROWT * (H3 / 16 / CT) / 8, 256, 0, stream>>>(
        msgf, M, Bih, b_ih, gi, H3, M / 32, ROWT);
    // gh = hidden @ W_hh^T + b_hh : [NV,256] x [256,768]
    gemm_wmma_kernel<float, CT><<<ROWT * (H3 / 16 / CT) / 8, 256, 0, stream>>>(
        catA, CAT, Bhh, b_hh, gh, H3, H / 32, ROWT);
    // GRU elementwise + refresh f16 hidden in catA
    gru_update_kernel<<<blocks(NV * H, 256), 256, 0, stream>>>(gi, gh, hidden, catA, nmask);
  }

  // --- readout GEMMs ---
  gemm_wmma_kernel<float, CT><<<ROWT * (OUTD / 16 / CT) / 8, 256, 0, stream>>>(
      catA, CAT, Bgatt, gab, gl, OUTD, CAT / 32, ROWT);    // gate logits [NV,320]x[320,256]
  gemm_wmma_kernel<float, CT><<<ROWT * (OUTD / 16 / CT) / 8, 256, 0, stream>>>(
      catA, CAT, Bgemb, geb, embp, OUTD, H / 32, ROWT);    // emb_out [NV,256]x[256,256]

  readout_kernel<<<GN, OUTD, 0, stream>>>(gl, embp, nmask, out);
}
